// NativeMoELayer_74036646248718
// MI455X (gfx1250) — matrix operand order repaired and checked by reading, not compiled, not run on hardware
//
#include <hip/hip_runtime.h>
#include <math.h>

// ---------------- problem constants ----------------
#define T_TOKENS 4096          // BATCH*SEQ
#define DMODEL   1024
#define HDIM     4096
#define PDIM     256
#define NEXP     8
#define CLAMP_MAX 4.6051701859880914f   // log(100)

#define TOK_TILE 16
#define HCHUNK   128
#define XSTRIDE  (DMODEL + 8)   // 1032 bf16 -> 2064B row, 16B aligned, bank step 4
#define HSTRIDE  (HCHUNK + 8)   // 136 bf16 -> 272B row, 16B aligned, bank step 4

typedef __attribute__((ext_vector_type(16))) __bf16 bf16x16;
typedef __attribute__((ext_vector_type(8)))  __bf16 bf16x8;
typedef __attribute__((ext_vector_type(8)))  float  f32x8;

// ---------------- helpers ----------------
__device__ __forceinline__ f32x8 wmma_bf16(bf16x16 a, bf16x16 b, f32x8 c) {
  // v_wmma_f32_16x16x32_bf16: (neg_a, A, neg_b, B, c_mod, C, reuse_a, reuse_b)
  return __builtin_amdgcn_wmma_f32_16x16x32_bf16(false, a, false, b, (short)0, c,
                                                 false, false);
}

// 16 consecutive f32 from global -> bf16x16 (fallback path, converts in-loop)
__device__ __forceinline__ bf16x16 load16f_bf16(const float* __restrict__ p) {
  const float4* q = (const float4*)p;
  float4 a = q[0], b = q[1], c = q[2], d = q[3];
  bf16x16 r;
  r[0]=(__bf16)a.x;  r[1]=(__bf16)a.y;  r[2]=(__bf16)a.z;  r[3]=(__bf16)a.w;
  r[4]=(__bf16)b.x;  r[5]=(__bf16)b.y;  r[6]=(__bf16)b.z;  r[7]=(__bf16)b.w;
  r[8]=(__bf16)c.x;  r[9]=(__bf16)c.y;  r[10]=(__bf16)c.z; r[11]=(__bf16)c.w;
  r[12]=(__bf16)d.x; r[13]=(__bf16)d.y; r[14]=(__bf16)d.z; r[15]=(__bf16)d.w;
  return r;
}

// A-operand frag from LDS: lane row holds contiguous K run {kb..kb+15}
__device__ __forceinline__ bf16x16 lds_afrag(const __bf16* base) {
  bf16x8 lo = *(const bf16x8*)base;
  bf16x8 hi = *(const bf16x8*)(base + 8);
  bf16x16 r;
#pragma unroll
  for (int i = 0; i < 8; ++i) { r[i] = lo[i]; r[i + 8] = hi[i]; }
  return r;
}

// ---------------- kernels ----------------
__global__ void zero_kernel(float* __restrict__ out, int n) {
  int i = blockIdx.x * 256 + threadIdx.x;
  if (i < n) out[i] = 0.0f;
}

// one-pass f32 -> bf16 weight conversion (A then B, 8 elems/thread, 16B stores)
__global__ void convert_kernel(const float* __restrict__ A,
                               const float* __restrict__ B,
                               __bf16* __restrict__ Abf,
                               __bf16* __restrict__ Bbf) {
  const size_t half = (size_t)NEXP * HDIM * DMODEL;
  size_t i = ((size_t)blockIdx.x * 256 + threadIdx.x) * 8;
  const float* src;
  __bf16* dst;
  size_t off;
  if (i < half) { src = A; dst = Abf; off = i; }
  else          { src = B; dst = Bbf; off = i - half; }
  const float4* p = (const float4*)(src + off);
  float4 a = p[0], b = p[1];
  bf16x8 r;
  r[0]=(__bf16)a.x; r[1]=(__bf16)a.y; r[2]=(__bf16)a.z; r[3]=(__bf16)a.w;
  r[4]=(__bf16)b.x; r[5]=(__bf16)b.y; r[6]=(__bf16)b.z; r[7]=(__bf16)b.w;
  *(bf16x8*)(dst + off) = r;
}

__global__ void prep_kernel(const float* __restrict__ sim,
                            const float* __restrict__ temp,
                            float* __restrict__ simn,
                            float* __restrict__ scale,
                            int* __restrict__ cnt) {
  __shared__ float inv[NEXP];
  int tid = threadIdx.x;
  if (tid < NEXP) {
    float s = 0.0f;
    for (int p = 0; p < PDIM; ++p) { float v = sim[p * NEXP + tid]; s += v * v; }
    inv[tid] = 1.0f / fmaxf(sqrtf(s), 1e-12f);
    cnt[tid] = 0;
  }
  if (tid == 0) scale[0] = expf(fminf(temp[0], CLAMP_MAX));
  __syncthreads();
  for (int i = tid; i < PDIM * NEXP; i += 256) simn[i] = sim[i] * inv[i & (NEXP - 1)];
}

__global__ void gate_kernel(const float* __restrict__ x,
                            const float* __restrict__ Wp,
                            const float* __restrict__ bp,
                            const float* __restrict__ simn,
                            const float* __restrict__ scale_p,
                            float* __restrict__ w_buf,
                            float* __restrict__ probs,
                            int* __restrict__ cnt,
                            int* __restrict__ list) {
  __shared__ float sx[DMODEL];
  __shared__ float red[256];
  __shared__ float part[256 * NEXP];
  const int t = blockIdx.x, tid = threadIdx.x;

  for (int i = tid; i < DMODEL; i += 256) sx[i] = x[(size_t)t * DMODEL + i];
  __syncthreads();

  // proj[tid] = bp[tid] + x . Wp[tid,:]
  const float* wrow = Wp + (size_t)tid * DMODEL;
  float acc = bp[tid];
#pragma unroll 8
  for (int k = 0; k < DMODEL; ++k) acc = fmaf(sx[k], wrow[k], acc);

  red[tid] = acc * acc;
  __syncthreads();
  for (int s = 128; s > 0; s >>= 1) {
    if (tid < s) red[tid] += red[tid + s];
    __syncthreads();
  }
  float pn = acc / fmaxf(sqrtf(red[0]), 1e-12f);

#pragma unroll
  for (int e = 0; e < NEXP; ++e) part[tid * NEXP + e] = pn * simn[tid * NEXP + e];
  __syncthreads();
  for (int s = 128; s > 0; s >>= 1) {
    if (tid < s)
#pragma unroll
      for (int e = 0; e < NEXP; ++e) part[tid * NEXP + e] += part[(tid + s) * NEXP + e];
    __syncthreads();
  }

  if (tid == 0) {
    const float sc = scale_p[0];
    float lg[NEXP], mx = -1e30f;
#pragma unroll
    for (int e = 0; e < NEXP; ++e) { lg[e] = part[e] * sc; mx = fmaxf(mx, lg[e]); }
    float pr[NEXP], sum = 0.0f;
#pragma unroll
    for (int e = 0; e < NEXP; ++e) { pr[e] = expf(lg[e] - mx); sum += pr[e]; }
    float isum = 1.0f / sum;
#pragma unroll
    for (int e = 0; e < NEXP; ++e) { pr[e] *= isum; probs[t * NEXP + e] = pr[e]; }
    // top-2, lowest index wins ties (matches jax.lax.top_k)
    int i1 = 0; float v1 = pr[0];
    for (int e = 1; e < NEXP; ++e) if (pr[e] > v1) { v1 = pr[e]; i1 = e; }
    int i2 = (i1 == 0) ? 1 : 0; float v2 = pr[i2];
    for (int e = 0; e < NEXP; ++e) if (e != i1 && pr[e] > v2) { v2 = pr[e]; i2 = e; }
    float s12 = 1.0f / (v1 + v2 + 1e-8f);
#pragma unroll
    for (int e = 0; e < NEXP; ++e) w_buf[t * NEXP + e] = 0.0f;
    w_buf[t * NEXP + i1] = v1 * s12;
    w_buf[t * NEXP + i2] = v2 * s12;
    int p1 = atomicAdd(&cnt[i1], 1); list[i1 * T_TOKENS + p1] = t;
    int p2 = atomicAdd(&cnt[i2], 1); list[i2 * T_TOKENS + p2] = t;
  }
}

__global__ void reduce_kernel(const float* __restrict__ probs,
                              const int* __restrict__ cnt,
                              float* __restrict__ out_tail) {
  __shared__ float red[256 * NEXP];
  int tid = threadIdx.x;
  float a[NEXP];
#pragma unroll
  for (int e = 0; e < NEXP; ++e) a[e] = 0.0f;
  for (int t = tid; t < T_TOKENS; t += 256)
#pragma unroll
    for (int e = 0; e < NEXP; ++e) a[e] += probs[t * NEXP + e];
#pragma unroll
  for (int e = 0; e < NEXP; ++e) red[tid * NEXP + e] = a[e];
  __syncthreads();
  for (int s = 128; s > 0; s >>= 1) {
    if (tid < s)
#pragma unroll
      for (int e = 0; e < NEXP; ++e) red[tid * NEXP + e] += red[(tid + s) * NEXP + e];
    __syncthreads();
  }
  if (tid == 0) {
    float aux = 0.0f;
    for (int e = 0; e < NEXP; ++e) {
      float fr = red[e] * (1.0f / T_TOKENS);
      out_tail[1 + e] = fr;                       // frac
      float d = fr - 1.0f / NEXP;
      aux += d * d;
    }
    out_tail[0] = aux;                            // aux_loss
    for (int e = 0; e < NEXP; ++e) out_tail[9 + e] = (float)cnt[e];  // usage
  }
}

// ---- fused routed expert kernel: out[t] += w * (x[t] + B(Ax + a) + b) ----
// BF16W=true: weights pre-converted to bf16 (no cvt in loop, half the bytes)
template <bool BF16W>
__global__ void expert_kernel(const float* __restrict__ x,
                              const float* __restrict__ A,
                              const float* __restrict__ a_bias,
                              const float* __restrict__ Bw,
                              const float* __restrict__ b_bias,
                              const __bf16* __restrict__ Abf,
                              const __bf16* __restrict__ Bbf,
                              const float* __restrict__ w_buf,
                              const int* __restrict__ cnt,
                              const int* __restrict__ list,
                              float* __restrict__ out) {
  __shared__ __align__(16) __bf16 lds_x[TOK_TILE * XSTRIDE];
  __shared__ __align__(16) __bf16 lds_h[TOK_TILE * HSTRIDE];
  __shared__ int   s_tok[TOK_TILE];
  __shared__ float s_wt[TOK_TILE];

  const int e    = blockIdx.y;
  const int tile = blockIdx.x;
  const int tid  = threadIdx.x;
  const int ce   = cnt[e];
  const int base = tile * TOK_TILE;
  if (base >= ce) return;                         // uniform early-exit
  const int valid = min(TOK_TILE, ce - base);

  if (tid < TOK_TILE) {
    int g = (tid < valid) ? list[e * T_TOKENS + base + tid]
                          : list[e * T_TOKENS + base];
    s_tok[tid] = g;
    s_wt[tid]  = (tid < valid) ? w_buf[g * NEXP + e] : 0.0f;
  }
  __syncthreads();

  // gather token tile, f32 -> bf16, into LDS
  for (int i = tid; i < TOK_TILE * DMODEL; i += 256) {
    int r = i >> 10, c = i & (DMODEL - 1);
    lds_x[r * XSTRIDE + c] = (__bf16)x[(size_t)s_tok[r] * DMODEL + c];
  }
  __syncthreads();

  const int wave  = tid >> 5;
  const int lane  = tid & 31;
  const int lhalf = lane >> 4;        // 0: K-lo half, 1: K-hi half (contiguous k = 16*lhalf + s)
  const int lrow  = lane & 15;

  const float*  Ae  = A   + (size_t)e * HDIM * DMODEL;
  const float*  Be  = Bw  + (size_t)e * DMODEL * HDIM;
  const __bf16* Aeb = Abf + (size_t)e * HDIM * DMODEL;
  const __bf16* Beb = Bbf + (size_t)e * DMODEL * HDIM;

  f32x8 facc[8];                       // 16 tokens x 128 cols per wave
#pragma unroll
  for (int n2 = 0; n2 < 8; ++n2)
#pragma unroll
    for (int j = 0; j < 8; ++j) facc[n2][j] = 0.0f;

  for (int hc = 0; hc < HDIM; hc += HCHUNK) {
    // ---- stage 1: h[16 x 128] = x_tile @ Ae[hc:hc+128,:]^T  (K = 1024) ----
    // software-pipelined: prefetch (A,B) frags of step k+1 before WMMA of step k
    f32x8 hacc;
#pragma unroll
    for (int j = 0; j < 8; ++j) hacc[j] = 0.0f;
    const int hrow = hc + wave * 16 + lrow;
    const float*  arow  = Ae  + (size_t)hrow * DMODEL + (lhalf << 4);
    const __bf16* arowb = Aeb + (size_t)hrow * DMODEL + (lhalf << 4);
    const __bf16* axp   = lds_x + lrow * XSTRIDE + (lhalf << 4);

    bf16x16 af = lds_afrag(axp);
    bf16x16 bw;
    if (BF16W) bw = *(const bf16x16*)arowb;
    else       bw = load16f_bf16(arow);
#pragma unroll 4
    for (int k0 = 0; k0 < DMODEL - 32; k0 += 32) {
      bf16x16 af2 = lds_afrag(axp + k0 + 32);
      bf16x16 bw2;
      if (BF16W) bw2 = *(const bf16x16*)(arowb + k0 + 32);
      else       bw2 = load16f_bf16(arow + k0 + 32);
      hacc = wmma_bf16(af, bw, hacc);
      af = af2; bw = bw2;
    }
    hacc = wmma_bf16(af, bw, hacc);

    const float ab = a_bias[e * HDIM + hrow];

    __syncthreads();                   // prior stage-2 readers of lds_h done
    {
      __bf16* hp = lds_h + (lhalf << 3) * HSTRIDE + wave * 16 + lrow;
#pragma unroll
      for (int j = 0; j < 8; ++j) hp[j * HSTRIDE] = (__bf16)(hacc[j] + ab);
    }
    __syncthreads();

    // ---- stage 2: facc[16 x 1024] += h @ Be[:, hc:hc+128]^T  (K = 128) ----
    // A-frags of the whole chunk preloaded once; B-frags prefetched one ahead.
    bf16x16 haf[4];
#pragma unroll
    for (int i = 0; i < 4; ++i)
      haf[i] = lds_afrag(lds_h + lrow * HSTRIDE + i * 32 + (lhalf << 4));

    const float*  bbf32 = Be  + (size_t)(wave * 128 + lrow) * HDIM + hc + (lhalf << 4);
    const __bf16* bb16  = Beb + (size_t)(wave * 128 + lrow) * HDIM + hc + (lhalf << 4);

    bf16x16 bcur;
    if (BF16W) bcur = *(const bf16x16*)bb16;
    else       bcur = load16f_bf16(bbf32);
#pragma unroll
    for (int it = 0; it < 32; ++it) {          // it = k*8 + n2 (n2 inner)
      bf16x16 bnext = bcur;
      if (it != 31) {
        const int jn = (it + 1) & 7, jk = (it + 1) >> 3;
        if (BF16W) bnext = *(const bf16x16*)(bb16 + (size_t)jn * (16 * HDIM) + jk * 32);
        else       bnext = load16f_bf16(bbf32 + (size_t)jn * (16 * HDIM) + jk * 32);
      }
      facc[it & 7] = wmma_bf16(haf[it >> 3], bcur, facc[it & 7]);
      bcur = bnext;
    }
  }

  // ---- epilogue: out[t,d] += w_t * (x[t,d] + f + b_bias[d]) ----
  const float* bb = b_bias + e * DMODEL;
#pragma unroll
  for (int n2 = 0; n2 < 8; ++n2) {
    const int d = wave * 128 + n2 * 16 + lrow;
    const float bbv = bb[d];
#pragma unroll
    for (int j = 0; j < 8; ++j) {
      const int tl = j + (lhalf << 3);
      if (tl < valid) {
        const int g = s_tok[tl];
        const float val = s_wt[tl] * (x[(size_t)g * DMODEL + d] + facc[n2][j] + bbv);
        atomicAdd(out + (size_t)g * DMODEL + d, val);
      }
    }
  }
}

// ---------------- launch ----------------
extern "C" void kernel_launch(void* const* d_in, const int* in_sizes, int n_in,
                              void* d_out, int out_size, void* d_ws, size_t ws_size,
                              hipStream_t stream) {
  (void)in_sizes; (void)n_in; (void)out_size;
  const float* x    = (const float*)d_in[0];
  const float* Wp   = (const float*)d_in[1];
  const float* bp   = (const float*)d_in[2];
  const float* sim  = (const float*)d_in[3];
  const float* temp = (const float*)d_in[4];
  const float* A    = (const float*)d_in[5];
  const float* ab   = (const float*)d_in[6];
  const float* Bw   = (const float*)d_in[7];
  const float* bb   = (const float*)d_in[8];
  float* out = (float*)d_out;

  // workspace layout: ~402 KB gate scratch, then optional bf16 weight mirror
  float* ws_simn  = (float*)d_ws;                         // 2048
  float* ws_scale = ws_simn + PDIM * NEXP;                // 16 (1 used)
  int*   ws_cnt   = (int*)(ws_scale + 16);                // 16 (8 used)
  int*   ws_list  = ws_cnt + 16;                          // 8*4096
  float* ws_w     = (float*)(ws_list + NEXP * T_TOKENS);  // 4096*8
  float* ws_probs = ws_w + T_TOKENS * NEXP;               // 4096*8

  const size_t base_ws = 512 * 1024;                      // bytes reserved above
  const size_t whalf   = (size_t)NEXP * HDIM * DMODEL;    // elems per weight tensor
  __bf16* ws_Abf = (__bf16*)((char*)d_ws + base_ws);
  __bf16* ws_Bbf = ws_Abf + whalf;
  const bool pre = ws_size >= base_ws + 2 * whalf * sizeof(__bf16);  // constant/run

  const int n_out = T_TOKENS * DMODEL;                    // 4,194,304
  zero_kernel<<<n_out / 256, 256, 0, stream>>>(out, n_out);
  prep_kernel<<<1, 256, 0, stream>>>(sim, temp, ws_simn, ws_scale, ws_cnt);
  if (pre) {
    const int nblk = (int)(2 * whalf / 8 / 256);          // 32768
    convert_kernel<<<nblk, 256, 0, stream>>>(A, Bw, ws_Abf, ws_Bbf);
  }
  gate_kernel<<<T_TOKENS, 256, 0, stream>>>(x, Wp, bp, ws_simn, ws_scale,
                                            ws_w, ws_probs, ws_cnt, ws_list);
  reduce_kernel<<<1, 256, 0, stream>>>(ws_probs, ws_cnt, out + (size_t)n_out);
  dim3 eg(T_TOKENS / TOK_TILE, NEXP);
  if (pre)
    expert_kernel<true><<<eg, 256, 0, stream>>>(x, A, ab, Bw, bb, ws_Abf, ws_Bbf,
                                                ws_w, ws_cnt, ws_list, out);
  else
    expert_kernel<false><<<eg, 256, 0, stream>>>(x, A, ab, Bw, bb, nullptr, nullptr,
                                                 ws_w, ws_cnt, ws_list, out);
}
